// mLSTM_31044023616497
// MI455X (gfx1250) — matrix operand order repaired and checked by reading, not compiled, not run on hardware
//
#include <hip/hip_runtime.h>
#include <hip/hip_bf16.h>

// ---------------------------------------------------------------------------
// mLSTM block for gfx1250 (MI455X): all GEMM-shaped work on v_wmma_f32_16x16x32_bf16
// B=2, S=2048, E=512, NH=4, DH=128, KSZ=4
// ---------------------------------------------------------------------------

typedef __attribute__((ext_vector_type(16))) __bf16 bf16x16;
typedef __attribute__((ext_vector_type(8)))  __bf16 bf16x8;
typedef __attribute__((ext_vector_type(8)))  float  f32x8;

#define CB   2
#define CS   2048
#define CE   512
#define CNH  4
#define CDH  128

__device__ __forceinline__ bf16x16 bfcat(bf16x8 lo, bf16x8 hi) {
    return __builtin_shufflevector(lo, hi, 0,1,2,3,4,5,6,7,8,9,10,11,12,13,14,15);
}
__device__ __forceinline__ f32x8 wmma_bf16(bf16x16 a, bf16x16 b, f32x8 c) {
    // D = A*B + C, fp32 accumulate, 16x16x32 bf16
    return __builtin_amdgcn_wmma_f32_16x16x32_bf16(false, a, false, b, (short)0, c, false, false);
}
__device__ __forceinline__ f32x8 zero8() {
    f32x8 z;
#pragma unroll
    for (int i = 0; i < 8; ++i) z[i] = 0.f;
    return z;
}
// XOR swizzles on 8-element (16B) groups for conflict-free b128 LDS fragment reads
__device__ __forceinline__ int swz128(int r, int c) {
    return r * 128 + ((((c >> 3) ^ (r & 15)) << 3) | (c & 7));
}
__device__ __forceinline__ int swz64(int r, int c) {
    return r * 64 + ((((c >> 3) ^ (r & 7)) << 3) | (c & 7));
}

// ---------------------------------------------------------------------------
// Generic tiled GEMM:  C[m,n] = sum_k A[m,k]*B[n,k] (+bias[n])
// Tiles: 128(M) x 128(N), K-step 32. Block = 256 threads = 8 waves,
// wave w owns rows 16w..16w+15 of the M-tile across all 8 N sub-tiles.
// blockIdx.z batches independent GEMMs (per-head) via element strides.
// Fragments are staged in registers first so all ds_load_b128 are in flight
// before the back-to-back WMMA chain issues.
// ---------------------------------------------------------------------------
__launch_bounds__(256)
__global__ void gemm_wmma_kernel(const float* __restrict__ A, int lda, long aHS,
                                 const float* __restrict__ B, int ldb, long bHS,
                                 float* __restrict__ C, int ldc, long cHS,
                                 int K, const float* __restrict__ bias)
{
    __shared__ __bf16 as_[128 * 40];
    __shared__ __bf16 bs_[128 * 40];
    const int tid = threadIdx.x, lane = tid & 31, w = tid >> 5;
    const int n0 = blockIdx.x * 128, m0 = blockIdx.y * 128;
    A += (long)blockIdx.z * aHS;
    B += (long)blockIdx.z * bHS;
    C += (long)blockIdx.z * cHS;

    f32x8 acc[8];
#pragma unroll
    for (int t = 0; t < 8; ++t) acc[t] = zero8();

    for (int k0 = 0; k0 < K; k0 += 32) {
        __syncthreads();
        if (k0 + 32 < K) {
            __builtin_prefetch(A + (long)m0 * lda + k0 + 32, 0, 1);
            __builtin_prefetch(B + (long)n0 * ldb + k0 + 32, 0, 1);
        }
#pragma unroll
        for (int it = 0; it < 4; ++it) {
            int idx = it * 1024 + tid * 4;
            int r = idx >> 5, c = idx & 31;
            float4 av = *(const float4*)(A + (long)(m0 + r) * lda + (k0 + c));
            __bf16* pa = &as_[r * 40 + c];
            pa[0] = (__bf16)av.x; pa[1] = (__bf16)av.y;
            pa[2] = (__bf16)av.z; pa[3] = (__bf16)av.w;
            float4 bv = *(const float4*)(B + (long)(n0 + r) * ldb + (k0 + c));
            __bf16* pb = &bs_[r * 40 + c];
            pb[0] = (__bf16)bv.x; pb[1] = (__bf16)bv.y;
            pb[2] = (__bf16)bv.z; pb[3] = (__bf16)bv.w;
        }
        __syncthreads();
        // A fragment (16x32): lane<16 -> K 0-7 & 16-23, lane>=16 -> K 8-15 & 24-31
        int ar  = w * 16 + (lane & 15);
        int kbo = (lane & 16) ? 8 : 0;
        bf16x16 af = bfcat(*(const bf16x8*)&as_[ar * 40 + kbo],
                           *(const bf16x8*)&as_[ar * 40 + kbo + 16]);
        // stage all B fragments, then issue WMMAs back-to-back
        bf16x16 bfr[8];
#pragma unroll
        for (int t = 0; t < 8; ++t) {
            // B fragment (32x16): lane<16 -> K 0-7 & 16-23, lane>=16 -> K 8-15 & 24-31
            int br = t * 16 + (lane & 15);
            int Kb = (lane & 16) ? 8 : 0;
            bfr[t] = bfcat(*(const bf16x8*)&bs_[br * 40 + Kb],
                           *(const bf16x8*)&bs_[br * 40 + Kb + 16]);
        }
#pragma unroll
        for (int t = 0; t < 8; ++t)
            acc[t] = wmma_bf16(af, bfr[t], acc[t]);
    }
#pragma unroll
    for (int t = 0; t < 8; ++t) {
        int n = n0 + t * 16 + (lane & 15);
        float bb = bias ? bias[n] : 0.f;
#pragma unroll
        for (int v = 0; v < 8; ++v) {
            int m = m0 + w * 16 + v + ((lane & 16) ? 8 : 0);
            C[(long)m * ldc + n] = acc[t][v] + bb;
        }
    }
}

// ---------------------------------------------------------------------------
// Causal conv1d as GEMM over K = KSZ*E = 2048 against pre-transposed weights
// wt[n][kk*E + k], with shifted A rows (zero outside batch). Fused bias+SiLU.
// A = x_mlstm = x_inner[:, :512] (lda 1024). Output x_conv_act [4096,512].
// ---------------------------------------------------------------------------
__launch_bounds__(256)
__global__ void conv_gemm_kernel(const float* __restrict__ xin,  // [4096,1024]
                                 const float* __restrict__ wt,   // [512,2048]
                                 const float* __restrict__ cb,   // [512]
                                 float* __restrict__ out)        // [4096,512]
{
    __shared__ __bf16 as_[128 * 40];
    __shared__ __bf16 bs_[128 * 40];
    const int tid = threadIdx.x, lane = tid & 31, w = tid >> 5;
    const int n0 = blockIdx.x * 128, m0 = blockIdx.y * 128;
    const int batchStart = (m0 / CS) * CS;

    f32x8 acc[8];
#pragma unroll
    for (int t = 0; t < 8; ++t) acc[t] = zero8();

    for (int k0 = 0; k0 < 4 * CE; k0 += 32) {
        __syncthreads();
#pragma unroll
        for (int it = 0; it < 4; ++it) {
            int idx = it * 1024 + tid * 4;
            int r = idx >> 5, c = idx & 31;
            int Kg = k0 + c;
            int kk = Kg >> 9;           // which conv tap
            int kc = Kg & 511;          // input channel
            int srow = m0 + r + kk - 3; // causal shift
            float4 av;
            if (srow >= batchStart) {
                av = *(const float4*)(xin + (long)srow * 1024 + kc);
            } else {
                av.x = av.y = av.z = av.w = 0.f;
            }
            __bf16* pa = &as_[r * 40 + c];
            pa[0] = (__bf16)av.x; pa[1] = (__bf16)av.y;
            pa[2] = (__bf16)av.z; pa[3] = (__bf16)av.w;
            float4 bv = *(const float4*)(wt + (long)(n0 + r) * 2048 + Kg);
            __bf16* pb = &bs_[r * 40 + c];
            pb[0] = (__bf16)bv.x; pb[1] = (__bf16)bv.y;
            pb[2] = (__bf16)bv.z; pb[3] = (__bf16)bv.w;
        }
        __syncthreads();
        int ar  = w * 16 + (lane & 15);
        int kbo = (lane & 16) ? 8 : 0;
        bf16x16 af = bfcat(*(const bf16x8*)&as_[ar * 40 + kbo],
                           *(const bf16x8*)&as_[ar * 40 + kbo + 16]);
        bf16x16 bfr[8];
#pragma unroll
        for (int t = 0; t < 8; ++t) {
            int br = t * 16 + (lane & 15);
            int Kb = (lane & 16) ? 8 : 0;
            bfr[t] = bfcat(*(const bf16x8*)&bs_[br * 40 + Kb],
                           *(const bf16x8*)&bs_[br * 40 + Kb + 16]);
        }
#pragma unroll
        for (int t = 0; t < 8; ++t)
            acc[t] = wmma_bf16(af, bfr[t], acc[t]);
    }
#pragma unroll
    for (int t = 0; t < 8; ++t) {
        int n = n0 + t * 16 + (lane & 15);
        float bb = cb[n];
#pragma unroll
        for (int v = 0; v < 8; ++v) {
            int m = m0 + w * 16 + v + ((lane & 16) ? 8 : 0);
            float val = acc[t][v] + bb;
            float sg  = 1.f / (1.f + __expf(-val));
            out[(long)m * CE + n] = val * sg;    // SiLU
        }
    }
}

// conv_w [E][E][KSZ] -> wt [E][KSZ][E]
__global__ void transpose_convw_kernel(const float* __restrict__ cw,
                                       float* __restrict__ wt)
{
    int i = blockIdx.x * 256 + threadIdx.x;  // over 512*512*4
    int n  = i >> 11;
    int r  = i & 2047;
    int kc = r >> 2;
    int kk = r & 3;
    wt[(long)n * 2048 + kk * 512 + kc] = cw[i];
}

// ---------------------------------------------------------------------------
// Gates: igate/fgate[b,h,s] = [q|k|v][b,s,:] . W[h,:] + bias[h]
// ---------------------------------------------------------------------------
__launch_bounds__(128)
__global__ void gate_kernel(const float* __restrict__ qs, const float* __restrict__ ks,
                            const float* __restrict__ vs,
                            const float* __restrict__ Wi, const float* __restrict__ bi,
                            const float* __restrict__ Wf, const float* __restrict__ bfv,
                            float* __restrict__ igate, float* __restrict__ fgate)
{
    __shared__ float red[128];
    const int s = blockIdx.x;        // b*S + s
    const int tid = threadIdx.x;
    float accI[4] = {0.f, 0.f, 0.f, 0.f};
    float accF[4] = {0.f, 0.f, 0.f, 0.f};
    for (int c = tid; c < CE; c += 128) {
        float qv = qs[(long)s * CE + c];
        float kv = ks[(long)s * CE + c];
        float vv = vs[(long)s * CE + c];
#pragma unroll
        for (int h = 0; h < CNH; ++h) {
            const float* wi = Wi + h * (3 * CE);
            const float* wf = Wf + h * (3 * CE);
            accI[h] += qv * wi[c] + kv * wi[CE + c] + vv * wi[2 * CE + c];
            accF[h] += qv * wf[c] + kv * wf[CE + c] + vv * wf[2 * CE + c];
        }
    }
    const int b  = s / CS;
    const int ss = s & (CS - 1);
    for (int h = 0; h < CNH; ++h) {
        for (int pass = 0; pass < 2; ++pass) {
            __syncthreads();
            red[tid] = pass ? accF[h] : accI[h];
            __syncthreads();
            for (int off = 64; off > 0; off >>= 1) {
                if (tid < off) red[tid] += red[tid + off];
                __syncthreads();
            }
            if (tid == 0) {
                long o = (long)(b * CNH + h) * CS + ss;
                if (pass) fgate[o] = red[0] + bfv[h];
                else      igate[o] = red[0] + bi[h];
            }
        }
    }
}

// ---------------------------------------------------------------------------
// Per-(b,h) scan: lfc = cumsum(logsigmoid(f)); g = ig - lfc; gmax = cummax(g)
// ---------------------------------------------------------------------------
__launch_bounds__(256)
__global__ void scan_kernel(const float* __restrict__ igate, const float* __restrict__ fgate,
                            float* __restrict__ lfc, float* __restrict__ gval,
                            float* __restrict__ gmax)
{
    __shared__ float buf[CS];
    __shared__ float red[256];
    const int bh = blockIdx.x;
    const int tid = threadIdx.x;
    const float* fg = fgate + (long)bh * CS;
    const float* ig = igate + (long)bh * CS;

    // phase 1: cumsum of logsigmoid
    float loc[8];
    float run = 0.f;
#pragma unroll
    for (int e = 0; e < 8; ++e) {
        float x  = fg[tid * 8 + e];
        float ls = fminf(x, 0.f) - log1pf(__expf(-fabsf(x)));
        run += ls;
        loc[e] = run;
    }
    red[tid] = run;
    __syncthreads();
    if (tid == 0) {
        float r = 0.f;
        for (int i = 0; i < 256; ++i) { float t = red[i]; red[i] = r; r += t; }
    }
    __syncthreads();
    float off = red[tid];
#pragma unroll
    for (int e = 0; e < 8; ++e) {
        float v = loc[e] + off;
        buf[tid * 8 + e] = v;
        lfc[(long)bh * CS + tid * 8 + e] = v;
    }
    __syncthreads();
    // phase 2: g and running max
    float gm = -INFINITY;
    float gloc[8];
#pragma unroll
    for (int e = 0; e < 8; ++e) {
        float gv = ig[tid * 8 + e] - buf[tid * 8 + e];
        gval[(long)bh * CS + tid * 8 + e] = gv;
        gm = fmaxf(gm, gv);
        gloc[e] = gm;
    }
    red[tid] = gm;
    __syncthreads();
    if (tid == 0) {
        float r = -INFINITY;
        for (int i = 0; i < 256; ++i) { float t = red[i]; red[i] = r; r = fmaxf(r, t); }
    }
    __syncthreads();
    float moff = red[tid];
#pragma unroll
    for (int e = 0; e < 8; ++e)
        gmax[(long)bh * CS + tid * 8 + e] = fmaxf(gloc[e], moff);
}

// ---------------------------------------------------------------------------
// mLSTM attention: per (b,h), 128-row tiles, 64-col K/V tiles.
// D[i,j] = exp(g[j]-gmax[i]); C = (q.kT/sqrt(dh))*D; norm = max(|rowsum C|,
// exp(-lfc[i]-gmax[i])); H = C@V / (norm+1e-6).
// kbuf (swizzled 128x128 bf16) triple-duty: q staging -> K tiles -> P tiles.
// vbuf: transposed V tile (rows = dh, cols = j) for b128 B-fragment reads.
// ---------------------------------------------------------------------------
__launch_bounds__(256)
__global__ void attn_kernel(const float* __restrict__ qs, const float* __restrict__ ks,
                            const float* __restrict__ vs,
                            const float* __restrict__ lfc, const float* __restrict__ gval,
                            const float* __restrict__ gmax,
                            float* __restrict__ out)
{
    __shared__ __bf16 kbuf[128 * 128];
    __shared__ __bf16 vbuf[128 * 64];
    const int tid = threadIdx.x, lane = tid & 31, w = tid >> 5;
    const int i0 = blockIdx.x * 128;
    const int bh = blockIdx.y;
    const int b = bh / CNH, h = bh % CNH;
    const long gbase = ((long)b * CS) * CE + h * CDH;   // q/k/v element base
    const int  sbase = bh * CS;                          // stats base
    const float scale = 0.08838834764831845f;            // 1/sqrt(128)

    // stage q tile into kbuf (bf16, swizzled)
#pragma unroll
    for (int it = 0; it < 16; ++it) {
        int idx = it * 1024 + tid * 4;
        int r = idx >> 7, c = idx & 127;
        float4 xv = *(const float4*)(qs + gbase + (long)(i0 + r) * CE + c);
        int a = swz128(r, c);
        kbuf[a]     = (__bf16)xv.x; kbuf[a + 1] = (__bf16)xv.y;
        kbuf[a + 2] = (__bf16)xv.z; kbuf[a + 3] = (__bf16)xv.w;
    }
    __syncthreads();
    // q A-fragments for all 4 K-iters, kept in registers for the whole j-loop
    bf16x16 qf[4];
    {
        int r   = w * 16 + (lane & 15);
        int kbo = (lane & 16) ? 8 : 0;
#pragma unroll
        for (int kk = 0; kk < 4; ++kk) {
            int kb = kk * 32 + kbo;
            qf[kk] = bfcat(*(const bf16x8*)&kbuf[swz128(r, kb)],
                           *(const bf16x8*)&kbuf[swz128(r, kb + 16)]);
        }
    }
    // per-row stats in C-matrix lane layout
    float gmaxr[8], enr[8];
#pragma unroll
    for (int v = 0; v < 8; ++v) {
        int row = i0 + w * 16 + v + ((lane & 16) ? 8 : 0);
        float gm = gmax[sbase + row];
        gmaxr[v] = gm;
        enr[v]   = __expf(-(lfc[sbase + row] + gm));     // norm floor exp(-max_log_D)
    }
    f32x8 hacc[8];
#pragma unroll
    for (int t = 0; t < 8; ++t) hacc[t] = zero8();
    float rs[8];
#pragma unroll
    for (int v = 0; v < 8; ++v) rs[v] = 0.f;

    for (int j0 = 0; j0 < i0 + 128; j0 += 64) {
        __syncthreads();   // previous iteration's readers done with kbuf/vbuf
        if (j0 + 64 < i0 + 128) {
            __builtin_prefetch(ks + gbase + (long)(j0 + 64) * CE, 0, 1);
            __builtin_prefetch(vs + gbase + (long)(j0 + 64) * CE, 0, 1);
        }
        // K tile: rows j 0..63, cols dh 0..127
#pragma unroll
        for (int it = 0; it < 8; ++it) {
            int idx = it * 1024 + tid * 4;
            int jr = idx >> 7, c = idx & 127;
            float4 xv = *(const float4*)(ks + gbase + (long)(j0 + jr) * CE + c);
            int a = swz128(jr, c);
            kbuf[a]     = (__bf16)xv.x; kbuf[a + 1] = (__bf16)xv.y;
            kbuf[a + 2] = (__bf16)xv.z; kbuf[a + 3] = (__bf16)xv.w;
        }
        // V tile transposed: vbuf[d][j]; b32 loads keep the b16 stores ~2-way
#pragma unroll
        for (int it = 0; it < 32; ++it) {
            int idx = it * 256 + tid;
            int jr = idx >> 7, d = idx & 127;
            float xv = vs[gbase + (long)(j0 + jr) * CE + d];
            vbuf[swz64(d, jr)] = (__bf16)xv;
        }
        __syncthreads();
        // S = q @ kT over this 64-col tile: stage K fragments, then WMMA chain
        f32x8 sc[4];
#pragma unroll
        for (int t = 0; t < 4; ++t) sc[t] = zero8();
#pragma unroll
        for (int kk = 0; kk < 4; ++kk) {
            bf16x16 kf[4];
#pragma unroll
            for (int t = 0; t < 4; ++t) {
                int br = t * 16 + (lane & 15);
                int Kb = kk * 32 + ((lane & 16) ? 8 : 0);
                kf[t] = bfcat(*(const bf16x8*)&kbuf[swz128(br, Kb)],
                              *(const bf16x8*)&kbuf[swz128(br, Kb + 16)]);
            }
#pragma unroll
            for (int t = 0; t < 4; ++t)
                sc[t] = wmma_bf16(qf[kk], kf[t], sc[t]);
        }
        // gate weighting + causal mask -> P, accumulate row sums
#pragma unroll
        for (int t = 0; t < 4; ++t) {
            int jg = j0 + t * 16 + (lane & 15);
            float gj = gval[sbase + jg];
#pragma unroll
            for (int v = 0; v < 8; ++v) {
                int ii = i0 + w * 16 + v + ((lane & 16) ? 8 : 0);
                float p = (jg <= ii) ? sc[t][v] * scale * __expf(gj - gmaxr[v]) : 0.f;
                rs[v] += p;
                sc[t][v] = p;
            }
        }
        __syncthreads();   // all waves finished reading the K tile
        // write P (wave-private rows 16w..16w+15) into kbuf, cols 0..63
#pragma unroll
        for (int t = 0; t < 4; ++t) {
#pragma unroll
            for (int v = 0; v < 8; ++v) {
                int row = w * 16 + v + ((lane & 16) ? 8 : 0);
                int col = t * 16 + (lane & 15);
                kbuf[swz128(row, col)] = (__bf16)sc[t][v];
            }
        }
        // H += P @ V   (LDS ops in-order within a wave: safe to read back)
        {
            int r   = w * 16 + (lane & 15);
            int kbo = (lane & 16) ? 8 : 0;
#pragma unroll
            for (int kk = 0; kk < 2; ++kk) {
                int kb = kk * 32 + kbo;
                bf16x16 pf = bfcat(*(const bf16x8*)&kbuf[swz128(r, kb)],
                                   *(const bf16x8*)&kbuf[swz128(r, kb + 16)]);
                bf16x16 vf[8];
#pragma unroll
                for (int t = 0; t < 8; ++t) {
                    int br = t * 16 + (lane & 15);
                    int Kb = kk * 32 + ((lane & 16) ? 8 : 0);
                    vf[t] = bfcat(*(const bf16x8*)&vbuf[swz64(br, Kb)],
                                  *(const bf16x8*)&vbuf[swz64(br, Kb + 16)]);
                }
#pragma unroll
                for (int t = 0; t < 8; ++t)
                    hacc[t] = wmma_bf16(pf, vf[t], hacc[t]);
            }
        }
    }
    // normalize and store
#pragma unroll
    for (int v = 0; v < 8; ++v) {
        float r = rs[v];
        r += __shfl_xor(r, 1); r += __shfl_xor(r, 2);
        r += __shfl_xor(r, 4); r += __shfl_xor(r, 8);   // within 16-lane group
        float inv = 1.f / (fmaxf(fabsf(r), enr[v]) + 1e-6f);
        int row = i0 + w * 16 + v + ((lane & 16) ? 8 : 0);
#pragma unroll
        for (int t = 0; t < 8; ++t)
            out[gbase + (long)row * CE + t * 16 + (lane & 15)] = hacc[t][v] * inv;
    }
}

// ---------------------------------------------------------------------------
// Per-head group-norm + skip + SiLU(z) gating. Block = (b,s,h) slice of 128.
// ---------------------------------------------------------------------------
__launch_bounds__(128)
__global__ void norm_gate_kernel(const float* __restrict__ hattn,
                                 const float* __restrict__ xact,
                                 const float* __restrict__ xin,    // [.,1024], z at +512
                                 const float* __restrict__ skip,
                                 float* __restrict__ hstate)
{
    __shared__ float rsum[4], rsq[4];
    const int bs = blockIdx.x, h = blockIdx.y, t = threadIdx.x;
    const int e = h * CDH + t;
    float val = hattn[(long)bs * CE + e];
    float s = val, sq = val * val;
#pragma unroll
    for (int m = 1; m < 32; m <<= 1) {
        s  += __shfl_xor(s, m);
        sq += __shfl_xor(sq, m);
    }
    int wv = t >> 5, lane = t & 31;
    if (lane == 0) { rsum[wv] = s; rsq[wv] = sq; }
    __syncthreads();
    float S4  = rsum[0] + rsum[1] + rsum[2] + rsum[3];
    float SS4 = rsq[0] + rsq[1] + rsq[2] + rsq[3];
    float mean = S4 * (1.f / 128.f);
    float var  = SS4 * (1.f / 128.f) - mean * mean;
    float xn   = (val - mean) * rsqrtf(var + 1e-5f);
    float xa   = xact[(long)bs * CE + e];
    float z    = xin[(long)bs * 1024 + CE + e];
    float sg   = 1.f / (1.f + __expf(-z));
    hstate[(long)bs * CE + e] = (xn + skip[e] * xa) * (z * sg);
}

// ---------------------------------------------------------------------------
extern "C" void kernel_launch(void* const* d_in, const int* in_sizes, int n_in,
                              void* d_out, int out_size, void* d_ws, size_t ws_size,
                              hipStream_t stream)
{
    (void)in_sizes; (void)n_in; (void)out_size; (void)ws_size;
    const float* x      = (const float*)d_in[0];
    const float* W_up   = (const float*)d_in[1];
    const float* Wq     = (const float*)d_in[2];
    const float* Wk     = (const float*)d_in[3];
    const float* Wv     = (const float*)d_in[4];
    const float* conv_w = (const float*)d_in[5];
    const float* conv_b = (const float*)d_in[6];
    const float* Wi     = (const float*)d_in[7];
    const float* bi     = (const float*)d_in[8];
    const float* Wf     = (const float*)d_in[9];
    const float* bfv    = (const float*)d_in[10];
    const float* skip   = (const float*)d_in[11];
    const float* W_down = (const float*)d_in[12];
    const float* b_down = (const float*)d_in[13];
    float* out = (float*)d_out;

    const long M = (long)CB * CS;          // 4096
    float* ws      = (float*)d_ws;
    float* x_inner = ws;                       // [4096,1024]
    float* xact    = x_inner + M * 1024;       // [4096,512]
    float* qsb     = xact + M * CE;
    float* ksb     = qsb + M * CE;
    float* vsb     = ksb + M * CE;
    float* wt      = vsb + M * CE;             // [512,2048]
    float* hattn   = wt + (long)CE * 2048;     // [4096,512]
    float* hstate  = hattn + M * CE;           // [4096,512]
    float* igate   = hstate + M * CE;          // [B,NH,S]
    float* fgate   = igate + CB * CNH * CS;
    float* lfc     = fgate + CB * CNH * CS;
    float* gval    = lfc + CB * CNH * CS;
    float* gmaxa   = gval + CB * CNH * CS;

    // conv_w -> [E][KSZ][E]
    transpose_convw_kernel<<<4096, 256, 0, stream>>>(conv_w, wt);

    // x_inner = x @ W_up^T   [4096,1024]
    gemm_wmma_kernel<<<dim3(8, 32, 1), 256, 0, stream>>>(
        x, CE, 0, W_up, CE, 0, x_inner, 1024, 0, CE, nullptr);

    // x_conv_act = silu(conv1d(x_mlstm) + conv_b)
    conv_gemm_kernel<<<dim3(4, 32, 1), 256, 0, stream>>>(x_inner, wt, conv_b, xact);

    // headwise q, k (from x_conv_act) and v (from x_mlstm): per-head 128x128 GEMMs
    gemm_wmma_kernel<<<dim3(1, 32, CNH), 256, 0, stream>>>(
        xact, CE, CDH, Wq, CDH, CDH * CDH, qsb, CE, CDH, CDH, nullptr);
    gemm_wmma_kernel<<<dim3(1, 32, CNH), 256, 0, stream>>>(
        xact, CE, CDH, Wk, CDH, CDH * CDH, ksb, CE, CDH, CDH, nullptr);
    gemm_wmma_kernel<<<dim3(1, 32, CNH), 256, 0, stream>>>(
        x_inner, 1024, CDH, Wv, CDH, CDH * CDH, vsb, CE, CDH, CDH, nullptr);

    // input / forget gates
    gate_kernel<<<(int)M, 128, 0, stream>>>(qsb, ksb, vsb, Wi, bi, Wf, bfv, igate, fgate);

    // per-(b,h) log-sigmoid cumsum + running max
    scan_kernel<<<CB * CNH, 256, 0, stream>>>(igate, fgate, lfc, gval, gmaxa);

    // mLSTM attention
    attn_kernel<<<dim3(CS / 128, CB * CNH), 256, 0, stream>>>(
        qsb, ksb, vsb, lfc, gval, gmaxa, hattn);

    // group-norm + skip + silu(z) gating
    norm_gate_kernel<<<dim3((int)M, CNH), 128, 0, stream>>>(hattn, xact, x_inner, skip, hstate);

    // out = h_state @ W_down^T + b_down
    gemm_wmma_kernel<<<dim3(4, 32, 1), 256, 0, stream>>>(
        hstate, CE, 0, W_down, CE, 0, out, CE, 0, CE, b_down);
}
